// HyperLinear_38053410242727
// MI455X (gfx1250) — matrix operand order repaired and checked by reading, not compiled, not run on hardware
//
#include <hip/hip_runtime.h>
#include <math.h>

// ---------------- types for WMMA (CDNA5 / gfx1250, wave32) ----------------
typedef __bf16 bf16_t;
typedef __attribute__((ext_vector_type(8)))  bf16_t v8bf;
typedef __attribute__((ext_vector_type(16))) bf16_t v16bf;
typedef __attribute__((ext_vector_type(8)))  float  v8f;

#define B_  8
#define L_  4096
#define D_  512
#define C_  13
#define WV_ 128
#define P_  64
#define R_  64
#define N_  (C_ * P_)   // 832

__device__ __forceinline__ float geluf(float v) {
    return 0.5f * v * (1.0f + erff(v * 0.70710678118654752f));
}
__device__ __forceinline__ unsigned short f2bf(float f) {
    unsigned u = __builtin_bit_cast(unsigned, f);
    unsigned r = u + 0x7FFFu + ((u >> 16) & 1u);   // round-to-nearest-even
    return (unsigned short)(r >> 16);
}
__device__ __forceinline__ float bf2f(unsigned short h) {
    unsigned u = ((unsigned)h) << 16;
    return __builtin_bit_cast(float, u);
}

// ---------------- K1: x (f32) -> bf16 copy --------------------------------
__global__ __launch_bounds__(256) void k_cvt(const float* __restrict__ x,
                                             unsigned short* __restrict__ xb, int n) {
    int i = blockIdx.x * 256 + threadIdx.x;
    if (i < n) xb[i] = f2bf(x[i]);
}

// ---------------- K2: mean over tokens ------------------------------------
__global__ __launch_bounds__(256) void k_mean(const float* __restrict__ x,
                                              float* __restrict__ mean) {
    int t = blockIdx.x * 256 + threadIdx.x;           // 0..4095
    int b = t >> 9, d = t & 511;
    const float* p = x + (size_t)b * L_ * D_ + d;
    float s = 0.f;
    for (int l = 0; l < L_; ++l) s += p[(size_t)l * D_];
    mean[t] = s * (1.0f / (float)L_);
}

// ---------------- K3: q = mean@Wq ; wkq = (Wk@q) * H^-0.5 ------------------
__global__ __launch_bounds__(512) void k_qwkq(const float* __restrict__ mean,
                                              const float* __restrict__ Wq,
                                              const float* __restrict__ Wk,
                                              float* __restrict__ wkq) {
    __shared__ float qs[B_ * 64];
    int t = threadIdx.x;                               // 512
    {
        int b = t >> 6, h = t & 63;
        const float* m = mean + b * D_;
        float s = 0.f;
        for (int d = 0; d < D_; ++d) s += m[d] * Wq[d * 64 + h];
        qs[t] = s;
    }
    __syncthreads();
    for (int o = t; o < B_ * D_; o += 512) {
        int b = o >> 9, d = o & 511;
        const float* q = qs + b * 64;
        float s = 0.f;
        for (int h = 0; h < 64; ++h) s += Wk[d * 64 + h] * q[h];
        wkq[o] = s * 0.125f;                           // 1/sqrt(64)
    }
}

// ---------------- K4: scores[b,l] = x[b,l,:] . wkq[b,:] --------------------
__global__ __launch_bounds__(256) void k_scores(const unsigned short* __restrict__ xbf,
                                                const float* __restrict__ wkq,
                                                float* __restrict__ sc) {
    int g = blockIdx.x * 8 + (threadIdx.x >> 5);       // row id in [0, B*L)
    int lane = threadIdx.x & 31;
    int b = g >> 12;
    const unsigned short* p = xbf + (size_t)g * D_;
    const float* w = wkq + b * D_;
    float s = 0.f;
    for (int j = lane; j < D_; j += 32) s += bf2f(p[j]) * w[j];
    for (int o = 16; o > 0; o >>= 1) s += __shfl_xor(s, o, 32);
    if (lane == 0) sc[g] = s;
}

// ---------------- K5: softmax over L (per batch) ---------------------------
__global__ __launch_bounds__(1024) void k_softmax(float* __restrict__ sc) {
    __shared__ float red[1024];
    int b = blockIdx.x, t = threadIdx.x;
    float* s = sc + (size_t)b * L_;
    float mx = -3.4e38f;
    for (int j = t; j < L_; j += 1024) mx = fmaxf(mx, s[j]);
    red[t] = mx; __syncthreads();
    for (int o = 512; o > 0; o >>= 1) { if (t < o) red[t] = fmaxf(red[t], red[t + o]); __syncthreads(); }
    mx = red[0]; __syncthreads();
    float sum = 0.f;
    for (int j = t; j < L_; j += 1024) { float e = expf(s[j] - mx); s[j] = e; sum += e; }
    red[t] = sum; __syncthreads();
    for (int o = 512; o > 0; o >>= 1) { if (t < o) red[t] += red[t + o]; __syncthreads(); }
    float inv = 1.0f / red[0];
    for (int j = t; j < L_; j += 1024) s[j] *= inv;
}

// ---------------- K6: pooled[b,d] = sum_l attn * x --------------------------
__global__ __launch_bounds__(256) void k_pooled(const float* __restrict__ x,
                                                const float* __restrict__ attn,
                                                float* __restrict__ pooled) {
    int t = blockIdx.x * 256 + threadIdx.x;            // 0..4095
    int b = t >> 9, d = t & 511;
    const float* p = x + (size_t)b * L_ * D_ + d;
    const float* a = attn + (size_t)b * L_;
    float s = 0.f;
    for (int l = 0; l < L_; ++l) s += a[l] * p[(size_t)l * D_];
    pooled[t] = s;
}

// ---------------- K7: attended = pooled@Wv ; xfeat = fx2(gelu(fx1(...))) ----
__global__ __launch_bounds__(256) void k_attfeat(const float* __restrict__ pooled,
                                                 const float* __restrict__ Wv,
                                                 const float* __restrict__ fx1w,
                                                 const float* __restrict__ fx1b,
                                                 const float* __restrict__ fx2w,
                                                 const float* __restrict__ fx2b,
                                                 float* __restrict__ xfeat) {
    __shared__ float att[B_ * D_];    // 16 KB
    __shared__ float hb[B_ * WV_];    // 4 KB
    int t = threadIdx.x;
    for (int o = t; o < B_ * D_; o += 256) {
        int b = o >> 9, d = o & 511;
        const float* p = pooled + b * D_;
        float s = 0.f;
        for (int i = 0; i < D_; ++i) s += p[i] * Wv[i * D_ + d];
        att[o] = s;
    }
    __syncthreads();
    for (int o = t; o < B_ * WV_; o += 256) {
        int b = o >> 7, w = o & 127;
        const float* a = att + b * D_;
        float s = fx1b[w];
        for (int i = 0; i < D_; ++i) s += a[i] * fx1w[i * WV_ + w];
        hb[o] = geluf(s);
    }
    __syncthreads();
    for (int o = t; o < B_ * WV_; o += 256) {
        int b = o >> 7, w = o & 127;
        const float* h = hb + b * WV_;
        float s = fx2b[w];
        for (int i = 0; i < WV_; ++i) s += h[i] * fx2w[i * WV_ + w];
        xfeat[o] = s;
    }
}

// ---------------- K8: cross-modal fusion -> gen_in [B,C,WV] -----------------
__global__ __launch_bounds__(128) void k_fusion(const float* __restrict__ we,
                                                const float* __restrict__ xfeat,
                                                const float* __restrict__ m1w, const float* __restrict__ m1b,
                                                const float* __restrict__ m2w, const float* __restrict__ m2b,
                                                const float* __restrict__ fuw, const float* __restrict__ fub,
                                                const float* __restrict__ fug, const float* __restrict__ fubt,
                                                float* __restrict__ genin) {
    __shared__ float wr[128], xf[128], f1[128], f2[128], tv[128], r1[128], r2[128];
    int blk = blockIdx.x;                              // b*13 + c
    int b = blk / C_;
    int t = threadIdx.x;
    wr[t] = we[(size_t)blk * WV_ + t];
    xf[t] = xfeat[b * WV_ + t];
    __syncthreads();
    {
        float s1 = m1b[t], s2 = m2b[t];
        for (int i = 0; i < WV_; ++i) { s1 += wr[i] * m1w[i * WV_ + t]; s2 += xf[i] * m2w[i * WV_ + t]; }
        f1[t] = s1; f2[t] = s2;
    }
    __syncthreads();
    {
        float s = fub[t];
        for (int i = 0; i < WV_; ++i) s += f1[i] * fuw[i * WV_ + t];
        for (int i = 0; i < WV_; ++i) s += f2[i] * fuw[(WV_ + i) * WV_ + t];
        tv[t] = s; r1[t] = s; r2[t] = s * s;
    }
    __syncthreads();
    for (int o = 64; o > 0; o >>= 1) { if (t < o) { r1[t] += r1[t + o]; r2[t] += r2[t + o]; } __syncthreads(); }
    float mu = r1[0] * (1.0f / 128.0f);
    float var = r2[0] * (1.0f / 128.0f) - mu * mu;
    float fused = geluf((tv[t] - mu) * rsqrtf(var + 1e-5f) * fug[t] + fubt[t]);
    genin[(size_t)blk * WV_ + t] = fused + f1[t] + xf[t];
}

// ---------------- K9: generators -> bias + M^T (bf16) -----------------------
// Per (b,c): U[64x64], V[64x512]; MT[b][n=p*13+c][d] = sum_r U[p,r]*V[r,d] (bf16)
__global__ __launch_bounds__(256) void k_genmt(const float* __restrict__ genin,
    const float* __restrict__ ulg, const float* __restrict__ ulb,
    const float* __restrict__ u1w, const float* __restrict__ u1b,
    const float* __restrict__ u2w, const float* __restrict__ u2b,
    const float* __restrict__ u3w, const float* __restrict__ u3b,
    const float* __restrict__ vlg, const float* __restrict__ vlb,
    const float* __restrict__ v1w, const float* __restrict__ v1b,
    const float* __restrict__ v2w, const float* __restrict__ v2b,
    const float* __restrict__ v3w, const float* __restrict__ v3b,
    const float* __restrict__ blg, const float* __restrict__ blb,
    const float* __restrict__ b1w, const float* __restrict__ b1b,
    const float* __restrict__ b2w, const float* __restrict__ b2b,
    float* __restrict__ biasc, unsigned short* __restrict__ mt) {
    __shared__ float nrm[128], lnbuf[128], h1[128], h2u[64], h2v[64];
    __shared__ float Um[4096], Vt[4096];
    __shared__ float r1[256], r2[256];
    int blk = blockIdx.x;                              // b*13 + c
    int b = blk / C_, c = blk % C_;
    int t = threadIdx.x;

    float gv = (t < 128) ? genin[(size_t)blk * WV_ + t] : 0.f;
    r1[t] = gv; r2[t] = gv * gv; __syncthreads();
    for (int o = 128; o > 0; o >>= 1) { if (t < o) { r1[t] += r1[t + o]; r2[t] += r2[t + o]; } __syncthreads(); }
    float mu = r1[0] * (1.0f / 128.0f);
    float var = r2[0] * (1.0f / 128.0f) - mu * mu;
    float inv = rsqrtf(var + 1e-5f);
    if (t < 128) nrm[t] = (gv - mu) * inv;
    __syncthreads();

    // ---- U branch
    if (t < 128) lnbuf[t] = nrm[t] * ulg[t] + ulb[t];
    __syncthreads();
    if (t < 128) { float s = u1b[t]; for (int i = 0; i < 128; ++i) s += lnbuf[i] * u1w[i * 128 + t]; h1[t] = geluf(s); }
    __syncthreads();
    if (t < 64) { float s = u2b[t]; for (int i = 0; i < 128; ++i) s += h1[i] * u2w[i * 64 + t]; h2u[t] = geluf(s); }
    __syncthreads();
    for (int o = t; o < 4096; o += 256) {
        float s = u3b[o];
        for (int i = 0; i < 64; ++i) s += h2u[i] * u3w[i * 4096 + o];
        Um[o] = s;                                     // Um[p*64+r]
    }
    __syncthreads();

    // ---- V branch (only hidden h2v; V tiles materialized below)
    if (t < 128) lnbuf[t] = nrm[t] * vlg[t] + vlb[t];
    __syncthreads();
    if (t < 128) { float s = v1b[t]; for (int i = 0; i < 128; ++i) s += lnbuf[i] * v1w[i * 128 + t]; h1[t] = geluf(s); }
    __syncthreads();
    if (t < 64) { float s = v2b[t]; for (int i = 0; i < 128; ++i) s += h1[i] * v2w[i * 64 + t]; h2v[t] = geluf(s); }
    __syncthreads();

    // ---- bias branch
    if (t < 128) lnbuf[t] = nrm[t] * blg[t] + blb[t];
    __syncthreads();
    if (t < 64) { float s = b1b[t]; for (int i = 0; i < 128; ++i) s += lnbuf[i] * b1w[i * 64 + t]; h1[t] = geluf(s); }
    __syncthreads();
    if (t < 64) {
        float s = b2b[t];
        for (int i = 0; i < 64; ++i) s += h1[i] * b2w[i * 64 + t];
        biasc[b * N_ + t * C_ + c] = s;                // n = p*C + c
    }

    // ---- MT tiles: V tile (64 x 64d) then contract with U
    for (int d0 = 0; d0 < D_; d0 += 64) {
        __syncthreads();
        for (int o = t; o < 4096; o += 256) {
            int r = o >> 6, dd = o & 63;
            int oi = r * D_ + d0 + dd;
            float s = v3b[oi];
            for (int i = 0; i < 64; ++i) s += h2v[i] * v3w[i * 32768 + oi];
            Vt[o] = s;
        }
        __syncthreads();
        for (int o = t; o < 4096; o += 256) {
            int p = o >> 6, dd = o & 63;
            const float* up = Um + (p << 6);
            const float* vp = Vt + dd;
            float s = 0.f;
            for (int r = 0; r < 64; ++r) s += up[r] * vp[r << 6];
            mt[((size_t)b * N_ + (size_t)(p * C_ + c)) * D_ + d0 + dd] = f2bf(s);
        }
    }
}

// ---------------- K10: main WMMA GEMM: out[b] = Xbf[b] @ M[b] + bias --------
// grid(13, 32, 8): N-tile 64, M-tile 128, batch. 8 waves, wave = 16x64 strip.
__global__ __launch_bounds__(256) void k_gemm(const unsigned short* __restrict__ xbf,
                                              const unsigned short* __restrict__ mt,
                                              const float* __restrict__ biasc,
                                              float* __restrict__ out) {
    const int b    = blockIdx.z;
    const int m0   = blockIdx.y * 128 + (threadIdx.x >> 5) * 16;
    const int n0   = blockIdx.x * 64;
    const int lane = threadIdx.x & 31;
    const int lm   = lane & 15;
    const int h    = lane >> 4;

    const size_t arow = ((size_t)(b * L_ + m0 + lm)) * D_;
    v8f z = {};
    v8f acc[4] = {z, z, z, z};

    for (int k0 = 0; k0 < D_; k0 += 32) {
        // A fragment 16x32 bf16: lanes0-15 rows M, K = h*16 + {0..15}
        const v8bf* pa = (const v8bf*)(xbf + arow + k0 + h * 16);
        v16bf a = __builtin_shufflevector(pa[0], pa[1],
                                          0, 1, 2, 3, 4, 5, 6, 7,
                                          8, 9, 10, 11, 12, 13, 14, 15);
#pragma unroll
        for (int nf = 0; nf < 4; ++nf) {
            int n = n0 + nf * 16 + lm;
            // B fragment 32x16 bf16 from MT[n][k]: K = h*16 + {0..15}
            const v8bf* pb = (const v8bf*)(mt + ((size_t)(b * N_ + n)) * D_ + k0 + h * 16);
            v16bf bb = __builtin_shufflevector(pb[0], pb[1],
                                               0, 1, 2, 3, 4, 5, 6, 7,
                                               8, 9, 10, 11, 12, 13, 14, 15);
            acc[nf] = __builtin_amdgcn_wmma_f32_16x16x32_bf16(
                false, a, false, bb, (short)0, acc[nf], false, false);
        }
    }

#pragma unroll
    for (int nf = 0; nf < 4; ++nf) {
        int n = n0 + nf * 16 + lm;
        float bn = biasc[b * N_ + n];
#pragma unroll
        for (int r = 0; r < 8; ++r) {
            int m = m0 + r + 8 * h;                    // C/D layout: VGPR r, lane half h
            out[((size_t)(b * L_ + m)) * N_ + n] = acc[nf][r] + bn;
        }
    }
}

// ---------------- launcher --------------------------------------------------
extern "C" void kernel_launch(void* const* d_in, const int* in_sizes, int n_in,
                              void* d_out, int out_size, void* d_ws, size_t ws_size,
                              hipStream_t stream) {
    const float* x    = (const float*)d_in[0];
    const float* we   = (const float*)d_in[1];
    const float* Wq   = (const float*)d_in[2];
    const float* Wk   = (const float*)d_in[3];
    const float* Wv   = (const float*)d_in[4];
    const float* fx1w = (const float*)d_in[5];
    const float* fx1b = (const float*)d_in[6];
    const float* fx2w = (const float*)d_in[7];
    const float* fx2b = (const float*)d_in[8];
    const float* m1w  = (const float*)d_in[9];
    const float* m1b  = (const float*)d_in[10];
    const float* m2w  = (const float*)d_in[11];
    const float* m2b  = (const float*)d_in[12];
    const float* fuw  = (const float*)d_in[13];
    const float* fub  = (const float*)d_in[14];
    const float* fug  = (const float*)d_in[15];
    const float* fubt = (const float*)d_in[16];
    const float* ulg  = (const float*)d_in[17];
    const float* ulb  = (const float*)d_in[18];
    const float* u1w  = (const float*)d_in[19];
    const float* u1b  = (const float*)d_in[20];
    const float* u2w  = (const float*)d_in[21];
    const float* u2b  = (const float*)d_in[22];
    const float* u3w  = (const float*)d_in[23];
    const float* u3b  = (const float*)d_in[24];
    const float* vlg  = (const float*)d_in[25];
    const float* vlb  = (const float*)d_in[26];
    const float* v1w  = (const float*)d_in[27];
    const float* v1b  = (const float*)d_in[28];
    const float* v2w  = (const float*)d_in[29];
    const float* v2b  = (const float*)d_in[30];
    const float* v3w  = (const float*)d_in[31];
    const float* v3b  = (const float*)d_in[32];
    const float* blg  = (const float*)d_in[33];
    const float* blb  = (const float*)d_in[34];
    const float* b1w  = (const float*)d_in[35];
    const float* b1b  = (const float*)d_in[36];
    const float* b2w  = (const float*)d_in[37];
    const float* b2b  = (const float*)d_in[38];

    char* ws = (char*)d_ws;
    unsigned short* xbf  = (unsigned short*)(ws + 0);                 // 33,554,432 B
    unsigned short* mtb  = (unsigned short*)(ws + 33554432);          //  6,815,744 B
    float* mean   = (float*)(ws + 40370176);                          //     16,384 B
    float* wkq    = (float*)(ws + 40386560);                          //     16,384 B
    float* scores = (float*)(ws + 40402944);                          //    131,072 B
    float* pooled = (float*)(ws + 40534016);                          //     16,384 B
    float* xfeat  = (float*)(ws + 40550400);                          //      4,096 B
    float* genin  = (float*)(ws + 40554496);                          //     53,248 B
    float* biasc  = (float*)(ws + 40607744);                          //     26,624 B

    const int nx = B_ * L_ * D_;                                      // 16,777,216

    k_cvt    <<<(nx + 255) / 256, 256, 0, stream>>>(x, xbf, nx);
    k_mean   <<<16, 256, 0, stream>>>(x, mean);
    k_qwkq   <<<1, 512, 0, stream>>>(mean, Wq, Wk, wkq);
    k_scores <<<(B_ * L_) / 8, 256, 0, stream>>>(xbf, wkq, scores);
    k_softmax<<<B_, 1024, 0, stream>>>(scores);
    k_pooled <<<16, 256, 0, stream>>>(x, scores, pooled);
    k_attfeat<<<1, 256, 0, stream>>>(pooled, Wv, fx1w, fx1b, fx2w, fx2b, xfeat);
    k_fusion <<<B_ * C_, 128, 0, stream>>>(we, xfeat, m1w, m1b, m2w, m2b,
                                           fuw, fub, fug, fubt, genin);
    k_genmt  <<<B_ * C_, 256, 0, stream>>>(genin,
                                           ulg, ulb, u1w, u1b, u2w, u2b, u3w, u3b,
                                           vlg, vlb, v1w, v1b, v2w, v2b, v3w, v3b,
                                           blg, blb, b1w, b1b, b2w, b2b,
                                           biasc, mtb);
    dim3 gg(N_ / 64, L_ / 128, B_);
    k_gemm   <<<gg, 256, 0, stream>>>(xbf, mtb, biasc, (float*)d_out);
}
